// ResidualVectorQuantizer_56453050138861
// MI455X (gfx1250) — compile-verified, matrix operand order checked by reading, not run on hardware
//
#include <hip/hip_runtime.h>
#include <hip/hip_bf16.h>
#include <math.h>

typedef __attribute__((ext_vector_type(16))) _Float16 v16h;
typedef __attribute__((ext_vector_type(8)))  _Float16 v8h;
typedef __attribute__((ext_vector_type(8)))  float    v8f;
typedef __attribute__((ext_vector_type(4)))  int      v4i;

#define N_ROWS 32768
#define DIM    256
#define NE     256
#define KPAIRS 4096
#define NUMQ   4
#define BETA_F 0.25f
#define INV_TEMP 10.0f
#define EPS_N  1e-12f
#define CHUNKS 128            // column chunks for LSE GEMM (256 cols each)
#define COLS_PER_CHUNK 256

// d_out flat layout: x_q | mean_loss | outer_losses[4] | idx[N][4] (as float)
#define XQ_OFF    0
#define MEAN_OFF  (N_ROWS * DIM)
#define OUTER_OFF (MEAN_OFF + 1)
#define IDX_OFF   (OUTER_OFF + NUMQ)

// ---- CDNA5 async global->LDS path (ASYNCcnt), with guaranteed fallback ----
#define AS1 __attribute__((address_space(1)))
#define AS3 __attribute__((address_space(3)))

#if __has_builtin(__builtin_amdgcn_global_load_async_to_lds_b128)
#define HAVE_ASYNC_LDS 1
__device__ __forceinline__ void async_cp16(const _Float16* g, _Float16* l) {
  // global_load_async_to_lds_b128: per-lane 16B, LDS write tracked by ASYNCcnt
  // builtin signature (from probe diagnostic): (int4 global*, int4 shared*, imm, imm)
  __builtin_amdgcn_global_load_async_to_lds_b128(
      (AS1 v4i*)g, (AS3 v4i*)l, 0, 0);
}
__device__ __forceinline__ void wait_async0() {
#if __has_builtin(__builtin_amdgcn_s_wait_asynccnt)
  __builtin_amdgcn_s_wait_asynccnt(0);
#else
  asm volatile("s_wait_asynccnt 0x0" ::: "memory");
#endif
}
#else
#define HAVE_ASYNC_LDS 0
#endif

// Build a 16-bit WMMA A/B fragment from row-major memory (global or LDS).
// Caller passes p = base + (lane&15)*ld + k0 + ((lane>>4)<<3); fragment is
// two b128 loads: K[off..off+7] and K[off+16..off+23] (ISA 7.12.2 layout).
__device__ __forceinline__ v16h load_frag(const _Float16* p) {
  v8h a = *(const v8h*)(p);
  v8h b = *(const v8h*)(p + 16);
  v16h r;
#pragma unroll
  for (int i = 0; i < 8; ++i) { r[i] = a[i]; r[i + 8] = b[i]; }
  return r;
}

__device__ __forceinline__ void lse_comb(float& m, float& s, float m2, float s2) {
  float mn = fmaxf(m, m2);
  s = s * __expf(m - mn) + s2 * __expf(m2 - mn);
  m = mn;
}

// ---------- residual copy / accumulator init ----------
__global__ void copy_x(const float* __restrict__ x, float* __restrict__ residual) {
  size_t i = (size_t)blockIdx.x * blockDim.x + threadIdx.x;
  residual[i] = x[i];
}

__global__ void zero_acc(float* accs) {
  if (threadIdx.x < 8) accs[threadIdx.x] = 0.f;
}

// ---------- per-stage: L2-normalize rows -> f16, keep ||r|| ----------
__global__ void norm_rows(const float* __restrict__ residual,
                          _Float16* __restrict__ xn,
                          float* __restrict__ rownorm) {
  int gw = blockIdx.x * (blockDim.x >> 5) + (threadIdx.x >> 5);
  int lane = threadIdx.x & 31;
  if (gw >= N_ROWS) return;
  const float* r = residual + (size_t)gw * DIM;
  float rv[8]; float ss = 0.f;
#pragma unroll
  for (int t = 0; t < 8; ++t) { rv[t] = r[lane + t * 32]; ss += rv[t] * rv[t]; }
#pragma unroll
  for (int m = 1; m < 32; m <<= 1) ss += __shfl_xor(ss, m, 32);
  float inv = rsqrtf(ss + EPS_N);
  _Float16* o = xn + (size_t)gw * DIM;
#pragma unroll
  for (int t = 0; t < 8; ++t) o[lane + t * 32] = (_Float16)(rv[t] * inv);
  if (lane == 0) rownorm[gw] = sqrtf(ss + EPS_N);
}

// ---------- per-stage: gather xn[i1] rows, compute pos dot ----------
__global__ void pos_gather(const _Float16* __restrict__ xn,
                           const int* __restrict__ i1,
                           const int* __restrict__ i2,
                           _Float16* __restrict__ Agath,
                           float* __restrict__ pos) {
  int gw = blockIdx.x * (blockDim.x >> 5) + (threadIdx.x >> 5);
  int lane = threadIdx.x & 31;
  if (gw >= KPAIRS) return;
  const _Float16* a = xn + (size_t)i1[gw] * DIM;
  const _Float16* b = xn + (size_t)i2[gw] * DIM;
  _Float16* ao = Agath + (size_t)gw * DIM;
  float acc = 0.f;
#pragma unroll
  for (int t = 0; t < 8; ++t) {
    int d = lane + t * 32;
    _Float16 av = a[d];
    ao[d] = av;
    acc += (float)av * (float)b[d];
  }
#pragma unroll
  for (int m = 1; m < 32; m <<= 1) acc += __shfl_xor(acc, m, 32);
  if (lane == 0) pos[gw] = acc * INV_TEMP;
}

// ---------- per-stage: codebook -> f16 + ||e||^2 ----------
__global__ void cb_prep(const float* __restrict__ cb,
                        _Float16* __restrict__ cbh,
                        float* __restrict__ cbnorm) {
  int gw = blockIdx.x * (blockDim.x >> 5) + (threadIdx.x >> 5);
  int lane = threadIdx.x & 31;
  if (gw >= NE) return;
  const float* c = cb + (size_t)gw * DIM;
  float acc = 0.f;
#pragma unroll
  for (int t = 0; t < 8; ++t) {
    int d = lane + t * 32;
    float v = c[d];
    cbh[(size_t)gw * DIM + d] = (_Float16)v;
    acc += v * v;
  }
#pragma unroll
  for (int m = 1; m < 32; m <<= 1) acc += __shfl_xor(acc, m, 32);
  if (lane == 0) cbnorm[gw] = acc;
}

// ---------- per-stage: WMMA sim GEMM + online logsumexp partials ----------
// Block = 8 waves sharing one 256-col chunk: B subtiles staged in LDS once per
// block (8x less L2 traffic), double-buffered async prefetch when available.
__global__ void __launch_bounds__(256)
lse_gemm(const _Float16* __restrict__ Agath,
         const _Float16* __restrict__ xn,
         float* __restrict__ pm, float* __restrict__ ps) {
  __shared__ _Float16 Bsh[2][16 * DIM];   // 2 x 8 KB
  int waveId = threadIdx.x >> 5;
  int lane = threadIdx.x & 31;
  int group = blockIdx.x >> 7;            // 0..31 (256 row tiles / 8 waves)
  int chunk = blockIdx.x & (CHUNKS - 1);  // 0..127
  int rowTile = group * 8 + waveId;       // 0..255
  int colBase = chunk * COLS_PER_CHUNK;
  int roff = (lane & 15) * DIM + ((lane >> 4) << 3);
  int t = threadIdx.x;                    // staging: 2 x 16B per thread

  const _Float16* Ab = Agath + (size_t)rowTile * 16 * DIM + roff;
  v16h af[8];
#pragma unroll
  for (int kk = 0; kk < 8; ++kk) af[kk] = load_frag(Ab + kk * 32);

  float m[8], s[8];
#pragma unroll
  for (int j = 0; j < 8; ++j) { m[j] = -1e30f; s[j] = 0.f; }

  auto stage = [&](int ct, int b) {
    const _Float16* src = xn + (size_t)(colBase + ct * 16) * DIM;
#if HAVE_ASYNC_LDS
    async_cp16(src + t * 8,        &Bsh[b][t * 8]);
    async_cp16(src + 2048 + t * 8, &Bsh[b][2048 + t * 8]);
#else
    *(v8h*)&Bsh[b][t * 8]        = *(const v8h*)(src + t * 8);
    *(v8h*)&Bsh[b][2048 + t * 8] = *(const v8h*)(src + 2048 + t * 8);
#endif
  };
  auto stage_sync = [&]() {
#if HAVE_ASYNC_LDS
    wait_async0();
#endif
    __syncthreads();
  };

  stage(0, 0);
  stage_sync();

  for (int ct = 0; ct < 16; ++ct) {
    int b = ct & 1;
    if (ct < 15) stage(ct + 1, b ^ 1);    // prefetch next subtile
    const _Float16* Bb = &Bsh[b][0] + roff;
    v8f c = {};
#pragma unroll
    for (int kk = 0; kk < 8; ++kk) {
      v16h bf = load_frag(Bb + kk * 32);  // ds_load_b128 x2
      c = __builtin_amdgcn_wmma_f32_16x16x32_f16(false, af[kk], false, bf,
                                                 (short)0, c, false, false);
    }
#pragma unroll
    for (int j = 0; j < 8; ++j) {
      float v = c[j] * INV_TEMP;
      float mn = fmaxf(m[j], v);
      s[j] = s[j] * __expf(m[j] - mn) + __expf(v - mn);
      m[j] = mn;
    }
    if (ct < 15) stage_sync();
  }
  // combine across the 16 lanes sharing each row (half-wave xor reduce)
#pragma unroll
  for (int msk = 1; msk <= 8; msk <<= 1) {
#pragma unroll
    for (int j = 0; j < 8; ++j) {
      float m2 = __shfl_xor(m[j], msk, 32);
      float s2 = __shfl_xor(s[j], msk, 32);
      lse_comb(m[j], s[j], m2, s2);
    }
  }
  if ((lane & 15) == 0) {
    int rbase = rowTile * 16 + ((lane >> 4) << 3);   // rows 0..7 or 8..15
#pragma unroll
    for (int j = 0; j < 8; ++j) {
      pm[(size_t)(rbase + j) * CHUNKS + chunk] = m[j];
      ps[(size_t)(rbase + j) * CHUNKS + chunk] = s[j];
    }
  }
}

// ---------- per-stage: fold partials -> mean(LSE - pos) ----------
__global__ void lse_reduce(const float* __restrict__ pm,
                           const float* __restrict__ ps,
                           const float* __restrict__ pos,
                           float* __restrict__ outer_acc) {
  int gw = blockIdx.x * (blockDim.x >> 5) + (threadIdx.x >> 5);
  int lane = threadIdx.x & 31;
  if (gw >= KPAIRS) return;
  float m = -1e30f, s = 0.f;
#pragma unroll
  for (int t = 0; t < CHUNKS / 32; ++t) {
    int ch = lane + t * 32;
    lse_comb(m, s, pm[(size_t)gw * CHUNKS + ch], ps[(size_t)gw * CHUNKS + ch]);
  }
#pragma unroll
  for (int msk = 1; msk < 32; msk <<= 1) {
    float m2 = __shfl_xor(m, msk, 32);
    float s2 = __shfl_xor(s, msk, 32);
    lse_comb(m, s, m2, s2);
  }
  if (lane == 0) {
    float lse = m + __logf(s);
    atomicAdd(outer_acc, (lse - pos[gw]) * (1.0f / KPAIRS));
  }
}

// ---------- per-stage: WMMA nearest-code argmin (codebook via LDS) ----------
__global__ void __launch_bounds__(256)
quant_argmin(const _Float16* __restrict__ xn,
             const float* __restrict__ rownorm,
             const _Float16* __restrict__ cbh,
             const float* __restrict__ cbnorm,
             int* __restrict__ idxbuf) {
  __shared__ _Float16 Bsh[2][16 * DIM];
  int waveId = threadIdx.x >> 5;
  int lane = threadIdx.x & 31;
  int rowTile = blockIdx.x * 8 + waveId;      // 0..2047
  int roff = (lane & 15) * DIM + ((lane >> 4) << 3);
  int rsel = (lane >> 4) << 3;
  int t = threadIdx.x;

  const _Float16* Ab = xn + (size_t)rowTile * 16 * DIM + roff;
  v16h af[8];
#pragma unroll
  for (int kk = 0; kk < 8; ++kk) af[kk] = load_frag(Ab + kk * 32);

  float rn8[8];
#pragma unroll
  for (int j = 0; j < 8; ++j) rn8[j] = rownorm[rowTile * 16 + rsel + j];

  float bv[8]; int bi[8];
#pragma unroll
  for (int j = 0; j < 8; ++j) { bv[j] = 3.4e38f; bi[j] = 0; }

  auto stage = [&](int ct, int b) {
    const _Float16* src = cbh + (size_t)(ct * 16) * DIM;
#if HAVE_ASYNC_LDS
    async_cp16(src + t * 8,        &Bsh[b][t * 8]);
    async_cp16(src + 2048 + t * 8, &Bsh[b][2048 + t * 8]);
#else
    *(v8h*)&Bsh[b][t * 8]        = *(const v8h*)(src + t * 8);
    *(v8h*)&Bsh[b][2048 + t * 8] = *(const v8h*)(src + 2048 + t * 8);
#endif
  };
  auto stage_sync = [&]() {
#if HAVE_ASYNC_LDS
    wait_async0();
#endif
    __syncthreads();
  };

  stage(0, 0);
  stage_sync();

  for (int ct = 0; ct < 16; ++ct) {
    int b = ct & 1;
    if (ct < 15) stage(ct + 1, b ^ 1);
    int col = ct * 16 + (lane & 15);
    float cbn = cbnorm[col];
    const _Float16* Bb = &Bsh[b][0] + roff;
    v8f c = {};
#pragma unroll
    for (int kk = 0; kk < 8; ++kk) {
      v16h bf = load_frag(Bb + kk * 32);
      c = __builtin_amdgcn_wmma_f32_16x16x32_f16(false, af[kk], false, bf,
                                                 (short)0, c, false, false);
    }
#pragma unroll
    for (int j = 0; j < 8; ++j) {
      // ||e||^2 - 2 * ||r|| * (xn . e)   (row-constant ||r||^2 dropped)
      float sc = cbn - 2.f * rn8[j] * c[j];
      if (sc < bv[j]) { bv[j] = sc; bi[j] = col; }
    }
    if (ct < 15) stage_sync();
  }
#pragma unroll
  for (int msk = 1; msk <= 8; msk <<= 1) {
#pragma unroll
    for (int j = 0; j < 8; ++j) {
      float v2 = __shfl_xor(bv[j], msk, 32);
      int   i2 = __shfl_xor(bi[j], msk, 32);
      if (v2 < bv[j] || (v2 == bv[j] && i2 < bi[j])) { bv[j] = v2; bi[j] = i2; }
    }
  }
  if ((lane & 15) == 0) {
#pragma unroll
    for (int j = 0; j < 8; ++j) idxbuf[rowTile * 16 + rsel + j] = bi[j];
  }
}

// ---------- per-stage: residual update + vq loss + idx output ----------
__global__ void quant_update(float* __restrict__ residual,
                             const float* __restrict__ cb,
                             const int* __restrict__ idxbuf,
                             float* __restrict__ vq_acc,
                             float* __restrict__ out_idx,
                             int qstage) {
  int gw = blockIdx.x * (blockDim.x >> 5) + (threadIdx.x >> 5);
  int lane = threadIdx.x & 31;
  if (gw >= N_ROWS) return;
  int id = idxbuf[gw];
  const float* c = cb + (size_t)id * DIM;
  float* r = residual + (size_t)gw * DIM;
  float acc = 0.f;
#pragma unroll
  for (int t = 0; t < 8; ++t) {
    int d = lane + t * 32;
    float rv = r[d];
    float qv = c[d];
    float df = qv - rv;
    acc += df * df;
    r[d] = rv - qv;          // residual -= q
  }
#pragma unroll
  for (int m = 1; m < 32; m <<= 1) acc += __shfl_xor(acc, m, 32);
  if (lane == 0) {
    atomicAdd(vq_acc, acc);
    out_idx[(size_t)gw * NUMQ + qstage] = (float)id;
  }
}

// ---------- epilogue ----------
__global__ void write_xq(const float* __restrict__ x,
                         const float* __restrict__ residual,
                         float* __restrict__ xq) {
  size_t i = (size_t)blockIdx.x * blockDim.x + threadIdx.x;
  xq[i] = x[i] - residual[i];   // x_q = sum of q's = x - final residual
}

__global__ void finalize(const float* __restrict__ accs, float* __restrict__ out) {
  if (threadIdx.x == 0) {
    float ml = 0.f;
    for (int q = 0; q < NUMQ; ++q)
      ml += (1.f + BETA_F) * accs[4 + q] / ((float)N_ROWS * (float)DIM);
    out[MEAN_OFF] = ml / (float)NUMQ;
    for (int q = 0; q < NUMQ; ++q) out[OUTER_OFF + q] = accs[q];
  }
}

extern "C" void kernel_launch(void* const* d_in, const int* in_sizes, int n_in,
                              void* d_out, int out_size, void* d_ws, size_t ws_size,
                              hipStream_t stream) {
  (void)in_sizes; (void)n_in; (void)out_size; (void)ws_size;
  const float* x   = (const float*)d_in[0];
  const float* cbs = (const float*)d_in[1];
  const int*   i1  = (const int*)d_in[2];
  const int*   i2  = (const int*)d_in[3];
  float* out = (float*)d_out;

  // workspace carve-up (~57 MB, all 256B-aligned)
  char* w = (char*)d_ws;
  float*    residual = (float*)   (w);                 // 33554432 B
  _Float16* xn       = (_Float16*)(w + 33554432);      // 16777216 B
  float*    rownorm  = (float*)   (w + 50331648);      //   131072 B
  _Float16* Agath    = (_Float16*)(w + 50462720);      //  2097152 B
  float*    pos      = (float*)   (w + 52559872);      //    16384 B
  float*    pm       = (float*)   (w + 52576256);      //  2097152 B
  float*    ps       = (float*)   (w + 54673408);      //  2097152 B
  _Float16* cbh      = (_Float16*)(w + 56770560);      //   131072 B
  float*    cbnorm   = (float*)   (w + 56901632);      //     1024 B
  int*      idxbuf   = (int*)     (w + 56902656);      //   131072 B
  float*    accs     = (float*)   (w + 57033728);      // outer[4] | vq[4]

  copy_x  <<<(N_ROWS * DIM) / 256, 256, 0, stream>>>(x, residual);
  zero_acc<<<1, 32, 0, stream>>>(accs);

  for (int q = 0; q < NUMQ; ++q) {
    const float* cbq = cbs + (size_t)q * NE * DIM;
    norm_rows   <<<N_ROWS / 8, 256, 0, stream>>>(residual, xn, rownorm);
    pos_gather  <<<KPAIRS / 8, 256, 0, stream>>>(xn, i1, i2, Agath, pos);
    cb_prep     <<<NE / 8,     256, 0, stream>>>(cbq, cbh, cbnorm);
    lse_gemm    <<<(KPAIRS / 16 / 8) * CHUNKS, 256, 0, stream>>>(Agath, xn, pm, ps);
    lse_reduce  <<<KPAIRS / 8, 256, 0, stream>>>(pm, ps, pos, accs + q);
    quant_argmin<<<(N_ROWS / 16) / 8, 256, 0, stream>>>(xn, rownorm, cbh, cbnorm, idxbuf);
    quant_update<<<N_ROWS / 8, 256, 0, stream>>>(residual, cbq, idxbuf,
                                                 accs + 4 + q, out + IDX_OFF, q);
  }

  write_xq<<<(N_ROWS * DIM) / 256, 256, 0, stream>>>(x, residual, out + XQ_OFF);
  finalize<<<1, 32, 0, stream>>>(accs, out);
}
